// relu_linear_att_88476326297959
// MI455X (gfx1250) — matrix operand order repaired and checked by reading, not compiled, not run on hardware
//
#include <hip/hip_runtime.h>

// ---------------------------------------------------------------------------
// ReLU linear attention, MI455X (gfx1250, wave32, WMMA f32_16x16x32_f16)
//   q,k,v: [B=4, H=8, d=128, N=8192] f32.
//   Phase 1: VK[bh][129][128] = [v;1] @ relu(k)^T   (split-K over N, f32 atomics)
//   Phase 2: out = (VK[0:128] @ relu(q)) / (ksum . relu(q)), layout [B,N,H*128]
// Math in f16 (inputs are N(0,1); f16 mantissa >> needed), f32 accumulate.
// Roofline: 537 MB @ 23.3 TB/s ~ 23 us; 17.3 GFLOP trivially hidden by f16 WMMA.
// ---------------------------------------------------------------------------

typedef __attribute__((ext_vector_type(16))) _Float16 v16h;
typedef __attribute__((ext_vector_type(8)))  _Float16 h8;
typedef __attribute__((ext_vector_type(4)))  _Float16 h4;
typedef __attribute__((ext_vector_type(8)))  float    v8f;

#define NSEQ   8192
#define BH     32
#define SPLIT  8
#define NCHUNK (NSEQ / SPLIT)   // 1024
#define KBLK   32
#define LDSP   40               // phase-1 LDS row stride in halves (80 B: 16B-aligned, conflict-free)
#define AP     136              // phase-2 LDS row stride in halves (272 B: 16B-aligned, conflict-free)

__device__ __forceinline__ v16h mk16(h8 lo, h8 hi) {
    union { v16h v; h8 h[2]; } u;
    u.h[0] = lo; u.h[1] = hi;
    return u.v;
}

// ============================ Phase 1: VK = [v;1] @ relu(k)^T ===============
__global__ __launch_bounds__(256) void vk_kernel(const float* __restrict__ kmat,
                                                 const float* __restrict__ vmat,
                                                 float* __restrict__ vkws) {
    __shared__ _Float16 sV[128 * LDSP];   // A: v[d][n]      (f16)
    __shared__ _Float16 sK[128 * LDSP];   // B: relu(k)[e][n] (f16), row e contiguous in n

    const int bh   = blockIdx.x;
    const int tid  = threadIdx.x;
    const int lane = tid & 31;
    const int wave = tid >> 5;
    const size_t base = (size_t)bh * 128 * NSEQ;
    const int n0 = blockIdx.y * NCHUNK;

    v8f   acc[8] = {};   // wave owns M-tile = wave, 8 N'-tiles of 16x16 f32
    float ks = 0.f;      // threads 0..127: ksum[e = tid]

    for (int it = 0; it < NCHUNK / KBLK; ++it) {
        const int nb = n0 + it * KBLK;
        // Stage 128x32 of v and relu(k): 1024 float4 each, 4/thread, coalesced.
        #pragma unroll
        for (int i = 0; i < 4; ++i) {
            const int s   = i * 256 + tid;
            const int row = s >> 3;        // 8 float4 per 32-wide row
            const int c4  = (s & 7) * 4;
            const size_t g = base + (size_t)row * NSEQ + nb + c4;
            const float4 fv = *(const float4*)(vmat + g);
            const float4 fk = *(const float4*)(kmat + g);
            h4 hv, hk;
            hv.x = (_Float16)fv.x; hv.y = (_Float16)fv.y;
            hv.z = (_Float16)fv.z; hv.w = (_Float16)fv.w;
            hk.x = (_Float16)fmaxf(fk.x, 0.f); hk.y = (_Float16)fmaxf(fk.y, 0.f);
            hk.z = (_Float16)fmaxf(fk.z, 0.f); hk.w = (_Float16)fmaxf(fk.w, 0.f);
            *(h4*)(sV + row * LDSP + c4) = hv;
            *(h4*)(sK + row * LDSP + c4) = hk;
        }
        __syncthreads();

        // A fragment (16x32 f16): row m = 16*wave + lane%16; lanes 0-15 hold
        // K {0-7,16-23}, lanes 16-31 hold K {8-15,24-31}  (ISA 7.12.2).
        {
            const int m  = wave * 16 + (lane & 15);
            const int ko = (lane & 16) ? 8 : 0;
            const _Float16* ap = sV + m * LDSP;
            const v16h a = mk16(*(const h8*)(ap + ko), *(const h8*)(ap + ko + 16));
            #pragma unroll
            for (int e = 0; e < 8; ++e) {
                // B fragment (32x16): lane = column e, K split as A ({0-7,16-23} / {8-15,24-31}).
                const _Float16* bp = sK + (e * 16 + (lane & 15)) * LDSP + ((lane & 16) ? 8 : 0);
                const v16h b = mk16(*(const h8*)bp, *(const h8*)(bp + 16));
                acc[e] = __builtin_amdgcn_wmma_f32_16x16x32_f16(
                    false, a, false, b, (short)0, acc[e], false, false);
            }
        }
        if (tid < 128) {   // ksum row (VALU; tiny vs WMMA work)
            const _Float16* kr = sK + tid * LDSP;
            #pragma unroll
            for (int j = 0; j < KBLK; ++j) ks += (float)kr[j];
        }
        __syncthreads();
    }

    // C/D layout: element (M = r + 8*lane[4], N = lane[3:0]) in VGPR r.
    float* dst = vkws + (size_t)bh * 129 * 128;
    const int mrow = wave * 16 + 8 * (lane >> 4);
    const int ec   = lane & 15;
    #pragma unroll
    for (int e = 0; e < 8; ++e)
        #pragma unroll
        for (int r = 0; r < 8; ++r)
            atomicAdd(dst + (size_t)(mrow + r) * 128 + e * 16 + ec, acc[e][r]);
    if (tid < 128) atomicAdd(dst + 128 * 128 + tid, ks);
}

// ===================== Phase 2: out = (VK @ relu(q)) / denom ================
__global__ __launch_bounds__(256) void out_kernel(const float* __restrict__ qmat,
                                                  const float* __restrict__ vkws,
                                                  float* __restrict__ out) {
    __shared__ _Float16 sA [128 * AP];   // VK rows (f16)
    __shared__ _Float16 sQt[64 * AP];    // relu(q) transposed: [n][e]
    __shared__ float    sKs[128];
    __shared__ float    sDen[64];
    __shared__ float    sOut[64 * 128];  // [n][d] staging for coalesced output

    const int bh = blockIdx.x;
    const int b  = bh >> 3, h = bh & 7;
    const int nbase = blockIdx.y * 64;
    const int tid = threadIdx.x, lane = tid & 31, wave = tid >> 5;
    const float* vk = vkws + (size_t)bh * 129 * 128;

    // Stage A = VK[0:128][0:128] as f16: 4096 float4, 16/thread.
    #pragma unroll
    for (int i = 0; i < 16; ++i) {
        const int s = i * 256 + tid;
        const int row = s >> 5, c4 = (s & 31) * 4;
        const float4 f = *(const float4*)(vk + (size_t)row * 128 + c4);
        h4 hv;
        hv.x = (_Float16)f.x; hv.y = (_Float16)f.y;
        hv.z = (_Float16)f.z; hv.w = (_Float16)f.w;
        *(h4*)(sA + row * AP + c4) = hv;
    }
    if (tid < 128) sKs[tid] = vk[128 * 128 + tid];

    // Stage relu(q) transposed: 2048 float4, 8/thread (coalesced global reads).
    const size_t qb = (size_t)bh * 128 * NSEQ + nbase;
    #pragma unroll
    for (int i = 0; i < 8; ++i) {
        const int s = i * 256 + tid;
        const int e = s >> 4, c4 = (s & 15) * 4;
        const float4 f = *(const float4*)(qmat + qb + (size_t)e * NSEQ + c4);
        sQt[(c4 + 0) * AP + e] = (_Float16)fmaxf(f.x, 0.f);
        sQt[(c4 + 1) * AP + e] = (_Float16)fmaxf(f.y, 0.f);
        sQt[(c4 + 2) * AP + e] = (_Float16)fmaxf(f.z, 0.f);
        sQt[(c4 + 3) * AP + e] = (_Float16)fmaxf(f.w, 0.f);
    }
    __syncthreads();

    if (tid < 64) {   // denominator = ksum . relu(q[:,n])
        float d = 0.f;
        const _Float16* col = sQt + tid * AP;
        #pragma unroll 4
        for (int e = 0; e < 128; ++e) d += sKs[e] * (float)col[e];
        sDen[tid] = d;
    }

    // GEMM: wave owns M-tile = wave, 4 n-tiles; K = 128 in 4 WMMA steps.
    v8f acc[4] = {};
    const int m = wave * 16 + (lane & 15);
    #pragma unroll
    for (int k0 = 0; k0 < 128; k0 += 32) {
        const _Float16* ap = sA + m * AP + k0 + ((lane & 16) ? 8 : 0);
        const v16h a = mk16(*(const h8*)ap, *(const h8*)(ap + 16));
        #pragma unroll
        for (int nt = 0; nt < 4; ++nt) {
            const _Float16* bp = sQt + (nt * 16 + (lane & 15)) * AP + k0 + ((lane & 16) ? 8 : 0);
            const v16h bf = mk16(*(const h8*)bp, *(const h8*)(bp + 16));
            acc[nt] = __builtin_amdgcn_wmma_f32_16x16x32_f16(
                false, a, false, bf, (short)0, acc[nt], false, false);
        }
    }
    __syncthreads();   // sDen ready, sQt no longer needed

    const int mrow = wave * 16 + 8 * (lane >> 4);
    #pragma unroll
    for (int nt = 0; nt < 4; ++nt) {
        const int n = nt * 16 + (lane & 15);
        const float inv = 1.f / sDen[n];
        #pragma unroll
        for (int r = 0; r < 8; ++r)
            sOut[n * 128 + mrow + r] = acc[nt][r] * inv;
    }
    __syncthreads();

    // Coalesced store: out[b][nbase+n][h*128 + d], 2048 float4, 8/thread.
    #pragma unroll
    for (int i = 0; i < 8; ++i) {
        const int s = i * 256 + tid;
        const int n = s >> 5, c4 = (s & 31) * 4;
        const float4 f = *(const float4*)(sOut + n * 128 + c4);
        *(float4*)(out + ((size_t)b * NSEQ + nbase + n) * 1024 + h * 128 + c4) = f;
    }
}

// ===========================================================================
extern "C" void kernel_launch(void* const* d_in, const int* in_sizes, int n_in,
                              void* d_out, int out_size, void* d_ws, size_t ws_size,
                              hipStream_t stream) {
    const float* q = (const float*)d_in[0];
    const float* k = (const float*)d_in[1];
    const float* v = (const float*)d_in[2];
    float* out  = (float*)d_out;
    float* vkws = (float*)d_ws;   // [32][129][128] f32 = 2.1 MB

    hipMemsetAsync(d_ws, 0, (size_t)BH * 129 * 128 * sizeof(float), stream);
    vk_kernel <<<dim3(BH, SPLIT), 256, 0, stream>>>(k, v, vkws);
    out_kernel<<<dim3(BH, NSEQ / 64), 256, 0, stream>>>(q, vkws, out);
}
